// SelfAttention_420906795477
// MI455X (gfx1250) — matrix operand order repaired and checked by reading, not compiled, not run on hardware
//
#include <hip/hip_runtime.h>

typedef __attribute__((ext_vector_type(16))) _Float16 v16h;
typedef __attribute__((ext_vector_type(8)))  float    v8f;

#define BM 128
#define BN 256
#define BK 64
#define ALD 72   // BK + 8 pad; 72*2=144 B row stride (16B aligned)
#define BLD 72

// ---------------------------------------------------------------------------
// CDNA5 async global->LDS copy (16B per lane), tracked with ASYNCcnt.
// vdst = per-lane LDS byte offset (addrspace(3) value = low 32 bits of flat
// address per ISA aperture rule), vaddr = per-lane 64-bit global address.
// ---------------------------------------------------------------------------
__device__ __forceinline__ void async_cp16(unsigned lds_off, const void* g)
{
  asm volatile("global_load_async_to_lds_b128 %0, %1, off"
               :: "v"(lds_off), "v"((unsigned long long)g)
               : "memory");
}
#define ASYNC_WAIT_12() asm volatile("s_wait_asynccnt 12" ::: "memory")
#define ASYNC_WAIT_0()  asm volatile("s_wait_asynccnt 0"  ::: "memory")

// fragment loaders: two 16-byte LDS reads each -> ds_load_b128 pairs
__device__ __forceinline__ v16h load_frag_a(const _Float16* p) {
  // A 16x32 layout: elems 0..7 -> K = half*8 + j ; elems 8..15 -> K = 16 + half*8 + j
  union { v16h v; uint4 q[2]; } f;
  f.q[0] = *(const uint4*)(p);
  f.q[1] = *(const uint4*)(p + 16);
  return f.v;
}
__device__ __forceinline__ v16h load_frag_b(const _Float16* p) {
  // B 32x16 layout: elems 0..7 -> K = half*8 + j ; elems 8..15 -> K = 16 + half*8 + j
  union { v16h v; uint4 q[2]; } f;
  f.q[0] = *(const uint4*)(p);
  f.q[1] = *(const uint4*)(p + 16);
  return f.v;
}

// ---------------------------------------------------------------------------
// WMMA GEMM, C[z] = A[z] * B[z]^T-style operand:
//   A row-major [M][K] (lda), B stored [N][K] (ldb = K-dim stride)
//   OUT_MODE 0: float, row-major [M][N] (ldc, batch stride sC)
//   OUT_MODE 1: f16,   row-major [M][N]
//   OUT_MODE 2: f16, transposed + re-batched: out[(row>>11)*sC + col*ldc + (row&2047)]
// Block 256 threads = 8 waves (2 M x 4 N), wave tile 64x64, block tile 128x256,
// K-step 64.  Double-buffered LDS filled by async DMA (ASYNCcnt).
// ---------------------------------------------------------------------------
template<int OUT_MODE>
__global__ __launch_bounds__(256)
void wmma_gemm(const _Float16* __restrict__ Ag, const _Float16* __restrict__ Bg,
               void* __restrict__ Cg, int Kdim,
               int lda, int ldb, int ldc,
               long sA, long sB, long sC)
{
  __shared__ _Float16 As[2][BM * ALD];   // [m][k]
  __shared__ _Float16 Bs[2][BN * BLD];   // [n][k]

  const int z = blockIdx.z;
  const _Float16* A = Ag + (long)z * sA;
  const _Float16* B = Bg + (long)z * sB;

  const int n0 = blockIdx.x * BN;
  const int m0 = blockIdx.y * BM;

  const int tid  = threadIdx.x;
  const int wid  = tid >> 5;
  const int lane = tid & 31;
  const int wm   = (wid & 1) * 64;   // wave M offset in tile
  const int wn   = (wid >> 1) * 64;  // wave N offset in tile
  const int half = lane >> 4;
  const int l16  = lane & 15;

  // per-thread tile-load coordinates (4 A chunks + 8 B chunks of 16B)
  const int arow = tid >> 3;          // 0..31  (+32*i)
  const int akc  = (tid & 7) * 8;

  v8f acc[4][4];
  #pragma unroll
  for (int mt = 0; mt < 4; ++mt)
    #pragma unroll
    for (int nt = 0; nt < 4; ++nt)
      acc[mt][nt] = {};

  const int T = Kdim / BK;

  // prologue: DMA tile 0 into buffer 0
  #pragma unroll
  for (int i = 0; i < 4; ++i) {
    int row = arow + i * 32;
    async_cp16((unsigned)(size_t)&As[0][row * ALD + akc],
               A + (long)(m0 + row) * lda + akc);
  }
  #pragma unroll
  for (int i = 0; i < 8; ++i) {
    int n = arow + i * 32;
    async_cp16((unsigned)(size_t)&Bs[0][n * BLD + akc],
               B + (long)(n0 + n) * ldb + akc);
  }

  for (int it = 0; it < T; ++it) {
    if (it + 1 < T) {
      const int knext = (it + 1) * BK;
      const int nb    = (it + 1) & 1;
      #pragma unroll
      for (int i = 0; i < 4; ++i) {
        int row = arow + i * 32;
        async_cp16((unsigned)(size_t)&As[nb][row * ALD + akc],
                   A + (long)(m0 + row) * lda + knext + akc);
      }
      #pragma unroll
      for (int i = 0; i < 8; ++i) {
        int n = arow + i * 32;
        async_cp16((unsigned)(size_t)&Bs[nb][n * BLD + akc],
                   B + (long)(n0 + n) * ldb + knext + akc);
      }
      ASYNC_WAIT_12();   // tile `it` (12 older async ops) complete
    } else {
      ASYNC_WAIT_0();
    }
    __syncthreads();

    const _Float16* as = &As[it & 1][0];
    const _Float16* bs = &Bs[it & 1][0];
    #pragma unroll
    for (int kk = 0; kk < BK; kk += 32) {
      v16h bfrag[4];
      #pragma unroll
      for (int nt = 0; nt < 4; ++nt)
        bfrag[nt] = load_frag_b(&bs[(wn + nt * 16 + l16) * BLD + kk + half * 8]);
      #pragma unroll
      for (int mt = 0; mt < 4; ++mt) {
        v16h afrag = load_frag_a(&as[(wm + mt * 16 + l16) * ALD + kk + half * 8]);
        #pragma unroll
        for (int nt = 0; nt < 4; ++nt)
          acc[mt][nt] = __builtin_amdgcn_wmma_f32_16x16x32_f16(
              false, afrag, false, bfrag[nt], (short)0, acc[mt][nt],
              false, false);
      }
    }
    __syncthreads();   // protect buffer (it&1) before it is re-filled
  }

  // ---- epilogue; C/D layout: elem r -> (M = r + half*8, N = l16) ----
  #pragma unroll
  for (int mt = 0; mt < 4; ++mt) {
    #pragma unroll
    for (int nt = 0; nt < 4; ++nt) {
      const int col = n0 + wn + nt * 16 + l16;
      const int rb  = m0 + wm + mt * 16 + half * 8;   // 8 consecutive rows
      if constexpr (OUT_MODE == 2) {
        union { uint4 q; _Float16 h[8]; } pk;
        #pragma unroll
        for (int r = 0; r < 8; ++r) pk.h[r] = (_Float16)acc[mt][nt][r];
        _Float16* dst = (_Float16*)Cg + (long)(rb >> 11) * sC
                        + (long)col * ldc + (rb & 2047);
        *(uint4*)dst = pk.q;
      } else if constexpr (OUT_MODE == 1) {
        #pragma unroll
        for (int r = 0; r < 8; ++r)
          ((_Float16*)Cg)[(long)z * sC + (long)(rb + r) * ldc + col] =
              (_Float16)acc[mt][nt][r];
      } else {
        #pragma unroll
        for (int r = 0; r < 8; ++r)
          ((float*)Cg)[(long)z * sC + (long)(rb + r) * ldc + col] =
              acc[mt][nt][r];
      }
    }
  }
}

// ---------------------------------------------------------------------------
// f32 -> f16 elementwise
__global__ void f32_to_f16(const float* __restrict__ in,
                           _Float16* __restrict__ out, long n)
{
  long i      = (long)blockIdx.x * blockDim.x + threadIdx.x;
  long stride = (long)gridDim.x * blockDim.x;
  for (; i < n; i += stride) out[i] = (_Float16)in[i];
}

// f32 -> f16 with per-slice transpose: out[s][o][d] = in[s][d][o] * scale(s)
// (slice 0 scaled by 1/8 to fold the 1/sqrt(64) attention scale into Q)
__global__ void f32_to_f16_wT(const float* __restrict__ in,
                              _Float16* __restrict__ out, int D)
{
  long n      = 3L * D * D;
  long i      = (long)blockIdx.x * blockDim.x + threadIdx.x;
  long stride = (long)gridDim.x * blockDim.x;
  for (; i < n; i += stride) {
    long s = i / ((long)D * D);
    long r = i - s * (long)D * D;
    int  d = (int)(r / D);
    int  o = (int)(r - (long)d * D);
    float v = in[i];
    if (s == 0) v *= 0.125f;
    out[s * (long)D * D + (long)o * D + d] = (_Float16)v;
  }
}

// ---------------------------------------------------------------------------
// In-place row softmax over f16 logits; row length 2048, 256 threads,
// 8 contiguous f16 (one uint4) per thread.
__global__ __launch_bounds__(256)
void softmax_rows(_Float16* __restrict__ P)
{
  __shared__ float red[256];
  const int t = threadIdx.x;
  _Float16* p = P + (long)blockIdx.x * 2048 + t * 8;

  union { uint4 q; _Float16 h[8]; } u;
  u.q = *(const uint4*)p;

  float vals[8];
  float m = -3.0e38f;
  #pragma unroll
  for (int i = 0; i < 8; ++i) {
    vals[i] = (float)u.h[i];
    m = fmaxf(m, vals[i]);
  }
  red[t] = m;
  __syncthreads();
  for (int s = 128; s > 0; s >>= 1) {
    if (t < s) red[t] = fmaxf(red[t], red[t + s]);
    __syncthreads();
  }
  const float rowmax = red[0];
  __syncthreads();

  float sum = 0.f;
  #pragma unroll
  for (int i = 0; i < 8; ++i) {
    vals[i] = __expf(vals[i] - rowmax);
    sum += vals[i];
  }
  red[t] = sum;
  __syncthreads();
  for (int s = 128; s > 0; s >>= 1) {
    if (t < s) red[t] += red[t + s];
    __syncthreads();
  }
  const float inv = 1.0f / red[0];

  #pragma unroll
  for (int i = 0; i < 8; ++i) u.h[i] = (_Float16)(vals[i] * inv);
  *(uint4*)p = u.q;
}

// ---------------------------------------------------------------------------
extern "C" void kernel_launch(void* const* d_in, const int* in_sizes, int n_in,
                              void* d_out, int out_size, void* d_ws, size_t ws_size,
                              hipStream_t stream)
{
  (void)in_sizes; (void)n_in; (void)out_size; (void)ws_size;
  const float* x = (const float*)d_in[0];   // [8,2048,768]
  const float* w = (const float*)d_in[1];   // [3,768,768]
  float* out = (float*)d_out;               // [8,2048,768]

  const long Bb = 8, S = 2048, D = 768;
  const long M  = Bb * S;                   // 16384

  // workspace (f16), all offsets 16B aligned:
  _Float16* Xh = (_Float16*)d_ws;           // [M][D]
  _Float16* Wh = Xh + M * D;                // [3][D][D], TRANSPOSED per slice ([o][d])
  _Float16* Qh = Wh + 3 * D * D;            // [M][D]  (pre-scaled by 1/8)
  _Float16* Kh = Qh + M * D;                // [M][D]
  _Float16* Vt = Kh + M * D;                // [Bb][D][S]  (V transposed per batch)
  _Float16* Ph = Vt + M * D;                // [Bb][S][S]

  // 1) input conversions (score scale folded into W0; W transposed)
  f32_to_f16<<<2048, 256, 0, stream>>>(x, Xh, M * D);
  f32_to_f16_wT<<<512, 256, 0, stream>>>(w, Wh, (int)D);

  // 2) Q,K projection: [16384x768] x W^T[768x768], batched z=2
  wmma_gemm<1><<<dim3(D / BN, M / BM, 2), 256, 0, stream>>>(
      Xh, Wh, Qh, (int)D, (int)D, (int)D, (int)D,
      0L, D * D, M * D);

  // 3) V projection, epilogue writes V transposed per batch -> Vt[b][o][t]
  wmma_gemm<2><<<dim3(D / BN, M / BM, 1), 256, 0, stream>>>(
      Xh, Wh + 2 * D * D, Vt, (int)D, (int)D, (int)D, (int)S,
      0L, 0L, D * S);

  // 4) scores: S_b = Q_b * K_b^T (scale pre-applied), batched z=8
  wmma_gemm<1><<<dim3(S / BN, S / BM, 8), 256, 0, stream>>>(
      Qh, Kh, Ph, (int)D, (int)D, (int)D, (int)S,
      S * D, S * D, S * S);

  // 5) row softmax in place
  softmax_rows<<<(int)M, 256, 0, stream>>>(Ph);

  // 6) output: O_b = P_b * V_b, B operand is Vt[b] stored [o][t] -> f32
  wmma_gemm<0><<<dim3(D / BN, S / BM, 8), 256, 0, stream>>>(
      Ph, Vt, out, (int)S, (int)S, (int)S, (int)D,
      S * S, D * S, S * D);
}
